// MultiHeadAttention_65420941853425
// MI455X (gfx1250) — matrix operand order repaired and checked by reading, not compiled, not run on hardware
//
#include <hip/hip_runtime.h>
#include <hip/hip_bf16.h>

// Problem constants (from reference): B=8, S=1024, D_MODEL=1024, H=16, Dh=64
#define BATCH   8
#define SEQ     1024
#define DMODEL  1024
#define NHEADS  16
#define HDIM    64
#define ROWS    (BATCH * SEQ)          // 8192

typedef __attribute__((ext_vector_type(16))) __bf16 v16bf;
typedef __attribute__((ext_vector_type(8)))  __bf16 v8bf;
typedef __attribute__((ext_vector_type(8)))  float  v8f;

// ---- CDNA5 wave32 WMMA VGPR striping (ISA 7.12.2) -------------------------
// A 16x32 (MxK) 16-bit: lane%16 = M.  K offset = (e&7) + ((e>>3)<<4) + ((lane&16)>>1)
//   => per lane: 8 consecutive K at  koff, 8 consecutive K at koff+16, koff = (lane&16)>>1
// B 32x16 (KxN) 16-bit: lane%16 = N.  K offset = (e&7) + ((e>>3)<<4) + ((lane&16)>>1)
//   => per lane: 8 consecutive K at koff, 8 consecutive K at koff+16 (same as A)
// C/D 16x16 f32: VGPR r, lane l -> M = r + ((l&16)>>1), N = l&15.

__device__ __forceinline__ v8bf ld8(const __bf16* p) {
    return *(const v8bf*)p;                       // 16B vector load
}
__device__ __forceinline__ v16bf cat16(v8bf lo, v8bf hi) {
    return __builtin_shufflevector(lo, hi, 0,1,2,3,4,5,6,7,8,9,10,11,12,13,14,15);
}
// Build an A/B fragment from four contiguous float4 chunks (f32 -> bf16).
__device__ __forceinline__ v16bf cvt16(float4 f0, float4 f1, float4 f2, float4 f3) {
    v16bf r;
    r[ 0]=(__bf16)f0.x; r[ 1]=(__bf16)f0.y; r[ 2]=(__bf16)f0.z; r[ 3]=(__bf16)f0.w;
    r[ 4]=(__bf16)f1.x; r[ 5]=(__bf16)f1.y; r[ 6]=(__bf16)f1.z; r[ 7]=(__bf16)f1.w;
    r[ 8]=(__bf16)f2.x; r[ 9]=(__bf16)f2.y; r[10]=(__bf16)f2.z; r[11]=(__bf16)f2.w;
    r[12]=(__bf16)f3.x; r[13]=(__bf16)f3.y; r[14]=(__bf16)f3.z; r[15]=(__bf16)f3.w;
    return r;
}
#define WMMA_BF16(A, B, C) \
    __builtin_amdgcn_wmma_f32_16x16x32_bf16(false, (A), false, (B), (short)0, (C), false, false)

// A-fragment (f32 source): 8+8 consecutive floats at koff / koff+16
__device__ __forceinline__ v16bf load_a_f32(const float* row, int k0, int lane) {
    const int koff = (lane & 16) >> 1;
    float4 f0 = *(const float4*)(row + k0 + koff);
    float4 f1 = *(const float4*)(row + k0 + koff + 4);
    float4 f2 = *(const float4*)(row + k0 + koff + 16);
    float4 f3 = *(const float4*)(row + k0 + koff + 20);
    return cvt16(f0, f1, f2, f3);
}
// A-fragment (bf16 source)
__device__ __forceinline__ v16bf load_a_bf16(const __bf16* row, int k0, int lane) {
    const int koff = (lane & 16) >> 1;
    return cat16(ld8(row + k0 + koff), ld8(row + k0 + koff + 16));
}
// B-fragment (f32 source): 8+8 consecutive floats at koff / koff+16
__device__ __forceinline__ v16bf load_b_f32(const float* row, int k0, int lane) {
    const int koff = (lane & 16) >> 1;
    float4 f0 = *(const float4*)(row + k0 + koff);
    float4 f1 = *(const float4*)(row + k0 + koff + 4);
    float4 f2 = *(const float4*)(row + k0 + koff + 16);
    float4 f3 = *(const float4*)(row + k0 + koff + 20);
    return cvt16(f0, f1, f2, f3);
}
// B-fragment (bf16 source)
__device__ __forceinline__ v16bf load_b_bf16(const __bf16* row, int k0, int lane) {
    const int koff = (lane & 16) >> 1;
    return cat16(ld8(row + k0 + koff), ld8(row + k0 + koff + 16));
}

// ---------------------------------------------------------------------------
// Projection GEMM: Y = scale * (X(f32)[ROWS,D] @ W(f32)[D,D]^T + bias), bf16 out.
// Register-blocked 16x64 per wave (A fragment reused across 4 WMMAs).
// MODE 0: row-major [ROWS, DMODEL].  MODE 1: per-head transposed Vt[b,h][n][k].
// ---------------------------------------------------------------------------
template <int MODE>
__global__ __launch_bounds__(256) void proj_gemm_kernel(
    const float* __restrict__ X, const float* __restrict__ W,
    const float* __restrict__ bias, __bf16* __restrict__ Y, float scale)
{
    const int lane = threadIdx.x & 31;
    const int wave = threadIdx.x >> 5;
    const int id  = blockIdx.x * 8 + wave;      // 512 * 16 = 8192 waves
    const int ntg = id & 15;                    // 16 column groups of 64
    const int tm  = id >> 4;                    // 512 row strips of 16

    const float* __restrict__ arow = X + (size_t)(tm * 16 + (lane & 15)) * DMODEL;
    const float* __restrict__ brow[4];
#pragma unroll
    for (int i = 0; i < 4; ++i)
        brow[i] = W + (size_t)(ntg * 64 + i * 16 + (lane & 15)) * DMODEL;

    v8f acc[4] = {};
    for (int k0 = 0; k0 < DMODEL; k0 += 32) {
        __builtin_prefetch(arow + k0 + 128, 0, 0);
        v16bf a = load_a_f32(arow, k0, lane);
#pragma unroll
        for (int i = 0; i < 4; ++i) {
            v16bf b = load_b_f32(brow[i], k0, lane);
            acc[i] = WMMA_BF16(a, b, acc[i]);
        }
    }

#pragma unroll
    for (int i = 0; i < 4; ++i) {
        const int   n  = ntg * 64 + i * 16 + (lane & 15);
        const float bv = bias[n];
#pragma unroll
        for (int r = 0; r < 8; ++r) {
            const int m = tm * 16 + r + ((lane & 16) >> 1);
            const __bf16 v = (__bf16)((acc[i][r] + bv) * scale);
            if (MODE == 0) {
                Y[(size_t)m * DMODEL + n] = v;
            } else {
                // Vt[((b*16 + h)*64 + n%64)*SEQ + s]
                const int b_ = m >> 10, s_ = m & (SEQ - 1);
                const int h_ = n >> 6,  nn = n & (HDIM - 1);
                Y[(((size_t)b_ * NHEADS + h_) * HDIM + nn) * SEQ + s_] = v;
            }
        }
    }
}

// ---------------------------------------------------------------------------
// Fused scores + softmax. One block per (bh, 16-row strip).
// Scores strip [16 x 1024] f32 lives in LDS (stride 1028 to de-conflict the
// two half-wave row groups); softmax in LDS; attention written once.
// ---------------------------------------------------------------------------
#define SROW 1028
__global__ __launch_bounds__(256) void scores_softmax_kernel(
    const __bf16* __restrict__ Qs, const __bf16* __restrict__ Ks,
    float* __restrict__ attn)
{
    extern __shared__ float smem[];             // 16*SROW scores + 256 reduction
    float* sc  = smem;
    float* red = smem + 16 * SROW;

    const int t    = threadIdx.x;
    const int lane = t & 31;
    const int wave = t >> 5;
    const int bh = blockIdx.y;                  // 0..127
    const int b  = bh >> 4, h = bh & 15;
    const int tm = blockIdx.x;                  // 0..63

    // Q fragments for the whole strip (Dh=64 -> k-steps 0 and 32), loaded once.
    const __bf16* __restrict__ qrow =
        Qs + (size_t)(b * SEQ + tm * 16 + (lane & 15)) * DMODEL + h * HDIM;
    const v16bf aq0 = load_a_bf16(qrow, 0,  lane);
    const v16bf aq1 = load_a_bf16(qrow, 32, lane);

    // Each wave produces 8 n-tiles (columns wave*128 .. wave*128+127).
#pragma unroll
    for (int i = 0; i < 8; ++i) {
        const int nt = wave * 8 + i;
        const __bf16* __restrict__ krow =
            Ks + (size_t)(b * SEQ + nt * 16 + (lane & 15)) * DMODEL + h * HDIM;
        v8f acc = {};
        acc = WMMA_BF16(aq0, load_b_bf16(krow, 0,  lane), acc);
        acc = WMMA_BF16(aq1, load_b_bf16(krow, 32, lane), acc);
#pragma unroll
        for (int r = 0; r < 8; ++r)
            sc[(r + ((lane & 16) >> 1)) * SROW + nt * 16 + (lane & 15)] = acc[r];
    }
    __syncthreads();

    // Row softmax: row = t>>4; 16 threads per row, strided columns.
    const int row = t >> 4, seg = t & 15;
    float* prow = sc + row * SROW;

    float mx = -1e30f;
#pragma unroll 4
    for (int j = 0; j < 64; ++j) mx = fmaxf(mx, prow[seg + 16 * j]);
    red[t] = mx; __syncthreads();
    for (int s = 8; s > 0; s >>= 1) {
        if (seg < s) red[t] = fmaxf(red[t], red[t + s]);
        __syncthreads();
    }
    mx = red[row * 16];
    __syncthreads();

    float sum = 0.f;
#pragma unroll 4
    for (int j = 0; j < 64; ++j) {
        const int c = seg + 16 * j;
        const float e = __expf(prow[c] - mx);
        prow[c] = e;
        sum += e;
    }
    red[t] = sum; __syncthreads();
    for (int s = 8; s > 0; s >>= 1) {
        if (seg < s) red[t] += red[t + s];
        __syncthreads();
    }

    // Normalize + single coalesced write of the attention strip.
    float* __restrict__ out = attn + ((size_t)bh * SEQ + tm * 16) * SEQ;
    for (int i = t; i < 16 * 256; i += 256) {
        const int r = i >> 8, q = i & 255;
        const float inv = 1.0f / red[r * 16];
        const float* s4 = sc + r * SROW + q * 4;
        float4 v = { s4[0] * inv, s4[1] * inv, s4[2] * inv, s4[3] * inv };
        ((float4*)(out + (size_t)r * SEQ))[q] = v;
    }
}

// ---------------------------------------------------------------------------
// Context: ctx[b, m, h*64+n] = sum_k attn[bh][m][k] * V[b, k, h*64+n]
// V comes pre-transposed per head: Vt[b,h][n][k] (bf16) -> contiguous B loads.
// Register-blocked 16x64 per wave (whole head).
// ---------------------------------------------------------------------------
__global__ __launch_bounds__(256) void context_kernel(
    const float* __restrict__ attn, const __bf16* __restrict__ Vt,
    __bf16* __restrict__ ctx)
{
    const int lane = threadIdx.x & 31;
    const int wave = threadIdx.x >> 5;
    const int bh = blockIdx.y;
    const int b  = bh >> 4, h = bh & 15;
    const int tm = blockIdx.x * 8 + wave;       // 64 strips per (b,h)

    const float* __restrict__ arow =
        attn + ((size_t)bh * SEQ + tm * 16 + (lane & 15)) * SEQ;
    const __bf16* __restrict__ vrow[4];
#pragma unroll
    for (int i = 0; i < 4; ++i)
        vrow[i] = Vt + ((size_t)bh * HDIM + i * 16 + (lane & 15)) * SEQ;

    v8f acc[4] = {};
    for (int k0 = 0; k0 < SEQ; k0 += 32) {
        __builtin_prefetch(arow + k0 + 128, 0, 0);
        v16bf a = load_a_f32(arow, k0, lane);
#pragma unroll
        for (int i = 0; i < 4; ++i) {
            v16bf bm = load_b_bf16(vrow[i], k0, lane);
            acc[i] = WMMA_BF16(a, bm, acc[i]);
        }
    }

#pragma unroll
    for (int i = 0; i < 4; ++i) {
        const int n = h * HDIM + i * 16 + (lane & 15);
#pragma unroll
        for (int r = 0; r < 8; ++r) {
            const int m = b * SEQ + tm * 16 + r + ((lane & 16) >> 1);
            ctx[(size_t)m * DMODEL + n] = (__bf16)acc[i][r];
        }
    }
}

// ---------------------------------------------------------------------------
// Output GEMM: out(f32) = ctx(bf16)[ROWS,D] @ Wo(f32)^T + bo. 16x64 per wave.
// ---------------------------------------------------------------------------
__global__ __launch_bounds__(256) void out_gemm_kernel(
    const __bf16* __restrict__ Xbf, const float* __restrict__ W,
    const float* __restrict__ bias, float* __restrict__ Y)
{
    const int lane = threadIdx.x & 31;
    const int wave = threadIdx.x >> 5;
    const int id  = blockIdx.x * 8 + wave;
    const int ntg = id & 15;
    const int tm  = id >> 4;

    const __bf16* __restrict__ arow = Xbf + (size_t)(tm * 16 + (lane & 15)) * DMODEL;
    const float* __restrict__ brow[4];
#pragma unroll
    for (int i = 0; i < 4; ++i)
        brow[i] = W + (size_t)(ntg * 64 + i * 16 + (lane & 15)) * DMODEL;

    v8f acc[4] = {};
    for (int k0 = 0; k0 < DMODEL; k0 += 32) {
        __builtin_prefetch(arow + k0 + 128, 0, 0);
        v16bf a = load_a_bf16(arow, k0, lane);
#pragma unroll
        for (int i = 0; i < 4; ++i) {
            v16bf b = load_b_f32(brow[i], k0, lane);
            acc[i] = WMMA_BF16(a, b, acc[i]);
        }
    }

#pragma unroll
    for (int i = 0; i < 4; ++i) {
        const int   n  = ntg * 64 + i * 16 + (lane & 15);
        const float bv = bias[n];
#pragma unroll
        for (int r = 0; r < 8; ++r) {
            const int m = tm * 16 + r + ((lane & 16) >> 1);
            Y[(size_t)m * DMODEL + n] = acc[i][r] + bv;
        }
    }
}

// ---------------------------------------------------------------------------
extern "C" void kernel_launch(void* const* d_in, const int* in_sizes, int n_in,
                              void* d_out, int out_size, void* d_ws, size_t ws_size,
                              hipStream_t stream) {
    (void)in_sizes; (void)n_in; (void)out_size; (void)ws_size;

    const float* query = (const float*)d_in[0];
    const float* key_  = (const float*)d_in[1];
    const float* value = (const float*)d_in[2];
    const float* Wq = (const float*)d_in[3];  const float* bq = (const float*)d_in[4];
    const float* Wk = (const float*)d_in[5];  const float* bk = (const float*)d_in[6];
    const float* Wv = (const float*)d_in[7];  const float* bv = (const float*)d_in[8];
    const float* Wo = (const float*)d_in[9];  const float* bo = (const float*)d_in[10];

    float* out  = (float*)d_out;                                  // [B,S,D] f32
    float* attn = out + (size_t)BATCH * SEQ * DMODEL;             // [B,H,S,S] f32

    // Workspace: bf16 Qs, Ks, Vt(per-head transposed), ctx (4 * 16 MB)
    __bf16* Qs  = (__bf16*)d_ws;
    __bf16* Ks  = Qs + (size_t)ROWS * DMODEL;
    __bf16* Vt  = Ks + (size_t)ROWS * DMODEL;
    __bf16* ctx = Vt + (size_t)ROWS * DMODEL;

    const dim3 blk(256);
    const int  gemm_blocks = (ROWS / 16) * (DMODEL / 64) / 8;     // 1024

    // Q carries the 1/sqrt(HEAD_DIM) scale.
    proj_gemm_kernel<0><<<gemm_blocks, blk, 0, stream>>>(query, Wq, bq, Qs, 0.125f);
    proj_gemm_kernel<0><<<gemm_blocks, blk, 0, stream>>>(key_,  Wk, bk, Ks, 1.0f);
    proj_gemm_kernel<1><<<gemm_blocks, blk, 0, stream>>>(value, Wv, bv, Vt, 1.0f);

    const size_t smem_bytes = (16 * SROW + 256) * sizeof(float);  // ~66.8 KB
    scores_softmax_kernel<<<dim3(SEQ / 16, BATCH * NHEADS), blk, smem_bytes, stream>>>(
        Qs, Ks, attn);

    context_kernel<<<dim3(8, BATCH * NHEADS), blk, 0, stream>>>(attn, Vt, ctx);
    out_gemm_kernel<<<gemm_blocks, blk, 0, stream>>>(ctx, Wo, bo, out);
}
